// CrossAggregator_25615184953755
// MI455X (gfx1250) — matrix operand order repaired and checked by reading, not compiled, not run on hardware
//
#include <hip/hip_runtime.h>

typedef _Float16 v4h  __attribute__((ext_vector_type(4)));
typedef _Float16 v8h  __attribute__((ext_vector_type(8)));
typedef _Float16 v16h __attribute__((ext_vector_type(16)));
typedef float    v4f  __attribute__((ext_vector_type(4)));
typedef float    v8f  __attribute__((ext_vector_type(8)));
typedef int      v4i  __attribute__((vector_size(4 * sizeof(int))));

typedef __attribute__((address_space(1))) v4i* gptr_v4i;
typedef __attribute__((address_space(3))) v4i* lptr_v4i;

#define R_TOTAL 4096      // B*M rows
#define DD      128       // D and OUT
#define FAN     16512     // D*D + D
#define KSPLIT  8         // outer-product chunks: 128 total, 16 per split
#define NSLAB   (KSPLIT + 1)   // + 1 slab for the self-vector chunk
#define WSTRIDE 136       // padded LDS row stride (halves): 68 DW -> 4n bank rotation

#if defined(__gfx1250__) && __has_builtin(__builtin_amdgcn_global_load_async_to_lds_b128) && __has_builtin(__builtin_amdgcn_s_wait_asynccnt)
#define USE_ASYNC 1
#else
#define USE_ASYNC 0
#endif

// ---------------- prep: entity masked mean -> f16 ea / eb ----------------
__global__ __launch_bounds__(256) void entity_kernel(
    const float* __restrict__ nv, const float* __restrict__ mk,
    _Float16* __restrict__ eah, _Float16* __restrict__ ebh)
{
  int g  = blockIdx.x * 256 + threadIdx.x;   // [0, 4096*2*128)
  int d  = g & 127;
  int bt = g >> 7;                           // bm*2 + side, [0, 8192)
  const float* base = nv + (size_t)bt * 32 * 128 + d;
  const float* mp   = mk + (size_t)bt * 32;
  float sum = 0.f;
#pragma unroll
  for (int n = 0; n < 32; ++n)
    sum += __builtin_nontemporal_load(base + n * 128) * mp[n];
  sum *= (1.0f / 32.0f);
  _Float16 hv = (_Float16)sum;
  int bm = bt >> 1;
  if (bt & 1) ebh[bm * 128 + d] = hv;
  else        eah[bm * 128 + d] = hv;
}

// ---------------- prep: f32 -> f16 convert (4 elems/thread) ----------------
__global__ __launch_bounds__(256) void cvt_kernel(
    const float* __restrict__ src, _Float16* __restrict__ dst, int n4)
{
  int i = blockIdx.x * 256 + threadIdx.x;
  if (i < n4) {
    v4f v = ((const v4f*)src)[i];
    v4h o;
#pragma unroll
    for (int e = 0; e < 4; ++e) o[e] = (_Float16)v[e];
    ((v4h*)dst)[i] = o;
  }
}

// ---------------- self-vector chunk: partial[slab 8] = selfh @ W[:,0:128]^T ----------------
__global__ __launch_bounds__(256) void self_gemm_kernel(
    const _Float16* __restrict__ Wh,
    const _Float16* __restrict__ selfh,
    float* __restrict__ pout)
{
  __shared__ __align__(16) _Float16 sW0[128 * WSTRIDE];

  const int t = threadIdx.x, lane = t & 31, wave = t >> 5;
  const int rslot = wave >> 1, nh = wave & 1;
  const int n16 = lane & 15, h = lane >> 4;
  const int wgRow = blockIdx.x * 128;

  for (int sidx = t; sidx < 128 * 16; sidx += 256) {
    int o = sidx >> 4, sg = sidx & 15;
    *(v8h*)&sW0[o * WSTRIDE + sg * 8] = *(const v8h*)&Wh[(size_t)o * FAN + sg * 8];
  }

  const int rowT0 = wgRow + rslot * 32;
  const int obase = nh * 64;
  v16h sa[2][4];
#pragma unroll
  for (int rt = 0; rt < 2; ++rt) {
    int rr = rowT0 + rt * 16 + n16;
#pragma unroll
    for (int c = 0; c < 4; ++c) {
      v8h lo = *(const v8h*)&selfh[(size_t)rr * DD + c * 32 + h * 8];
      v8h hi = *(const v8h*)&selfh[(size_t)rr * DD + c * 32 + 16 + h * 8];
      sa[rt][c] = __builtin_shufflevector(lo, hi,
          0,1,2,3,4,5,6,7,8,9,10,11,12,13,14,15);
    }
  }
  __syncthreads();

  auto loadB = [&](v16h* bv, int c) {
#pragma unroll
    for (int ot = 0; ot < 4; ++ot) {
      const _Float16* bp = &sW0[(obase + ot * 16 + n16) * WSTRIDE + c * 32 + h * 8];
      v8h blo = *(const v8h*)bp;
      v8h bhi = *(const v8h*)(bp + 16);
      bv[ot] = __builtin_shufflevector(blo, bhi,
          0,1,2,3,4,5,6,7,8,9,10,11,12,13,14,15);
    }
  };

  v8f acc[2][4] = {};
  v16h b0[4], b1[4];
  loadB(b0, 0);
#pragma unroll
  for (int c = 0; c < 4; ++c) {
    v16h* bc = (c & 1) ? b1 : b0;
    v16h* bn = (c & 1) ? b0 : b1;
    if (c < 3) loadB(bn, c + 1);   // 8 independent ds_load_b128 in flight
#pragma unroll
    for (int ot = 0; ot < 4; ++ot) {
      acc[0][ot] = __builtin_amdgcn_wmma_f32_16x16x32_f16(
          false, sa[0][c], false, bc[ot], (short)0, acc[0][ot], false, false);
      acc[1][ot] = __builtin_amdgcn_wmma_f32_16x16x32_f16(
          false, sa[1][c], false, bc[ot], (short)0, acc[1][ot], false, false);
    }
  }

#pragma unroll
  for (int rt = 0; rt < 2; ++rt)
#pragma unroll
    for (int ot = 0; ot < 4; ++ot) {
      int col = obase + ot * 16 + n16;
#pragma unroll
      for (int p = 0; p < 8; ++p) {
        int row = rowT0 + rt * 16 + p + 8 * h;
        pout[(size_t)row * DD + col] = acc[rt][ot][p];
      }
    }
}

// ---------------- main GEMM: outer-product chunks i=0..127, branch-free pipeline ----------------
__global__ __launch_bounds__(256) void gemm_kernel(
    const _Float16* __restrict__ Wh,
    const _Float16* __restrict__ eah,
    const _Float16* __restrict__ ebh,
    float* __restrict__ partial)
{
  __shared__ __align__(16) _Float16 sW[2][128 * WSTRIDE];  // W chunk double buffer
  __shared__ __align__(16) _Float16 sEA[128 * WSTRIDE];    // ea rows for this WG

  const int t = threadIdx.x, lane = t & 31, wave = t >> 5;
  const int rslot = wave >> 1;          // 0..3, 32 rows each
  const int nh    = wave & 1;           // 0..1, 64 outs each
  const int n16   = lane & 15, h = lane >> 4;

  const int wgRow = blockIdx.x * 128;
  const int split = blockIdx.y;
  const int kc0 = split * (128 / KSPLIT);
  const int kc1 = kc0 + (128 / KSPLIT);

  // stage ea rows (128 x 128 halves) into padded LDS
  for (int sidx = t; sidx < 128 * 16; sidx += 256) {
    int r = sidx >> 4, sg = sidx & 15;
    *(v8h*)&sEA[r * WSTRIDE + sg * 8] =
        *(const v8h*)&eah[(size_t)(wgRow + r) * DD + sg * 8];
  }

  // eb in WMMA-A layout, resident for whole K loop
  const int rowT0 = wgRow + rslot * 32;
  v16h ebA[2][4];
#pragma unroll
  for (int rt = 0; rt < 2; ++rt) {
    int rr = rowT0 + rt * 16 + n16;
#pragma unroll
    for (int c = 0; c < 4; ++c) {
      v8h lo = *(const v8h*)&ebh[(size_t)rr * DD + c * 32 + h * 8];
      v8h hi = *(const v8h*)&ebh[(size_t)rr * DD + c * 32 + 16 + h * 8];
      ebA[rt][c] = __builtin_shufflevector(lo, hi,
          0,1,2,3,4,5,6,7,8,9,10,11,12,13,14,15);
    }
  }

#if USE_ASYNC
  auto issueAsyncW = [&](int kc, int buf) {
    size_t colb = (size_t)(kc + 1) * 128;   // W column base for chunk i=kc
#pragma unroll
    for (int rpt = 0; rpt < 8; ++rpt) {
      int sidx = rpt * 256 + t;
      int o = sidx >> 4, sg = sidx & 15;
      __builtin_amdgcn_global_load_async_to_lds_b128(
          (gptr_v4i)(Wh + (size_t)o * FAN + colb + sg * 8),
          (lptr_v4i)(&sW[buf][o * WSTRIDE + sg * 8]),
          0, 0);
    }
  };
#else
  auto loadWregs = [&](int kc, v8h* pre) {
    size_t colb = (size_t)(kc + 1) * 128;
#pragma unroll
    for (int rpt = 0; rpt < 8; ++rpt) {
      int sidx = rpt * 256 + t;
      int o = sidx >> 4, sg = sidx & 15;
      pre[rpt] = *(const v8h*)&Wh[(size_t)o * FAN + colb + sg * 8];
    }
  };
  auto storeWregs = [&](int buf, const v8h* pre) {
#pragma unroll
    for (int rpt = 0; rpt < 8; ++rpt) {
      int sidx = rpt * 256 + t;
      int o = sidx >> 4, sg = sidx & 15;
      *(v8h*)&sW[buf][o * WSTRIDE + sg * 8] = pre[rpt];
    }
  };
#endif

  v8f acc[2][4] = {};   // [row-tile][o-tile]

#if USE_ASYNC
  issueAsyncW(kc0, 0);
#else
  {
    v8h pre[8];
    loadWregs(kc0, pre);
    storeWregs(0, pre);
  }
#endif

  const int obase = nh * 64;

  auto loadB = [&](v16h* bv, int buf, int c) {
#pragma unroll
    for (int ot = 0; ot < 4; ++ot) {
      const _Float16* bp = &sW[buf][(obase + ot * 16 + n16) * WSTRIDE + c * 32 + h * 8];
      v8h blo = *(const v8h*)bp;
      v8h bhi = *(const v8h*)(bp + 16);
      bv[ot] = __builtin_shufflevector(blo, bhi,
          0,1,2,3,4,5,6,7,8,9,10,11,12,13,14,15);
    }
  };

  for (int kc = kc0; kc < kc1; ++kc) {
    const int buf = (kc - kc0) & 1;
#if USE_ASYNC
    __builtin_amdgcn_s_wait_asynccnt(0);   // my async batch for sW[buf] has landed
#endif
    __syncthreads();                       // everyone's batch landed; buf^1 free
#if USE_ASYNC
    if (kc + 1 < kc1) issueAsyncW(kc + 1, buf ^ 1);
#else
    v8h pre[8];
    const bool more = (kc + 1 < kc1);
    if (more) loadWregs(kc + 1, pre);
#endif

    const _Float16 s0 = sEA[(rslot * 32 + n16) * WSTRIDE + kc];
    const _Float16 s1 = sEA[(rslot * 32 + 16 + n16) * WSTRIDE + kc];

    v16h b0[4], b1[4];
    loadB(b0, buf, 0);
#pragma unroll
    for (int c = 0; c < 4; ++c) {
      v16h* bc = (c & 1) ? b1 : b0;
      v16h* bn = (c & 1) ? b0 : b1;
      if (c < 3) loadB(bn, buf, c + 1);   // next 8 ds_load_b128 in flight over WMMAs
      // A row r = ea[r, kc] * eb[r, :] -> v_pk_mul_f16, co-executes with WMMA
      v16h sv0, sv1;
#pragma unroll
      for (int e = 0; e < 16; ++e) { sv0[e] = s0; sv1[e] = s1; }
      v16h a0 = sv0 * ebA[0][c];
      v16h a1 = sv1 * ebA[1][c];
#pragma unroll
      for (int ot = 0; ot < 4; ++ot) {
        acc[0][ot] = __builtin_amdgcn_wmma_f32_16x16x32_f16(
            false, a0, false, bc[ot], (short)0, acc[0][ot], false, false);
        acc[1][ot] = __builtin_amdgcn_wmma_f32_16x16x32_f16(
            false, a1, false, bc[ot], (short)0, acc[1][ot], false, false);
      }
    }
#if !USE_ASYNC
    if (more) storeWregs(buf ^ 1, pre);
#endif
  }

  // deterministic split-K: plain stores to this split's private slab
  float* pout = partial + (size_t)split * (R_TOTAL * DD);
#pragma unroll
  for (int rt = 0; rt < 2; ++rt)
#pragma unroll
    for (int ot = 0; ot < 4; ++ot) {
      int col = obase + ot * 16 + n16;
#pragma unroll
      for (int p = 0; p < 8; ++p) {
        int row = rowT0 + rt * 16 + p + 8 * h;   // C layout: M = vgpr + 8*(lane>=16)
        pout[(size_t)row * DD + col] = acc[rt][ot][p];
      }
    }
}

// ---------------- split-K reduce + bias ----------------
__global__ __launch_bounds__(256) void reduce_kernel(
    const float* __restrict__ partial, const float* __restrict__ bvec,
    float* __restrict__ out)
{
  int i = blockIdx.x * 256 + threadIdx.x;      // < 4096*128
  float v = bvec[i & 127];
#pragma unroll
  for (int s = 0; s < NSLAB; ++s)
    v += partial[(size_t)s * (R_TOTAL * DD) + i];
  out[i] = v;
}

extern "C" void kernel_launch(void* const* d_in, const int* in_sizes, int n_in,
                              void* d_out, int out_size, void* d_ws, size_t ws_size,
                              hipStream_t stream) {
  const float* d_self = (const float*)d_in[0];   // (4096, 128)
  const float* d_nv   = (const float*)d_in[1];   // (4096, 2, 32, 128)
  const float* d_mk   = (const float*)d_in[2];   // (4096, 2, 32)
  const float* d_W    = (const float*)d_in[3];   // (128, 16512)
  const float* d_b    = (const float*)d_in[4];   // (128,)
  float* out = (float*)d_out;

  char* ws = (char*)d_ws;
  _Float16* Wh      = (_Float16*)(ws);                 // 4,227,072 B
  _Float16* eah     = (_Float16*)(ws + 4227072);       // 1 MB
  _Float16* ebh     = (_Float16*)(ws + 5275648);       // 1 MB
  _Float16* selfh   = (_Float16*)(ws + 6324224);       // 1 MB
  float*    partial = (float*)(ws + 7372800);          // 9 * 2 MB

  cvt_kernel<<<(528384 + 255) / 256, 256, 0, stream>>>(d_W, Wh, 528384);
  cvt_kernel<<<(131072 + 255) / 256, 256, 0, stream>>>(d_self, selfh, 131072);
  entity_kernel<<<4096, 256, 0, stream>>>(d_nv, d_mk, eah, ebh);

  self_gemm_kernel<<<R_TOTAL / 128, 256, 0, stream>>>(
      Wh, selfh, partial + (size_t)KSPLIT * (R_TOTAL * DD));

  dim3 grid(R_TOTAL / 128, KSPLIT);
  gemm_kernel<<<grid, 256, 0, stream>>>(Wh, eah, ebh, partial);

  reduce_kernel<<<(R_TOTAL * DD) / 256, 256, 0, stream>>>(partial, d_b, out);
}
